// SingleHeadAttention_84310208020601
// MI455X (gfx1250) — compile-verified, matrix operand order checked by reading, not run on hardware
//
#include <hip/hip_runtime.h>

// Single-head causal attention for MI455X (gfx1250, wave32, WMMA, async-to-LDS).
// bf16x3 split-precision GEMMs on v_wmma_f32_16x16x32_bf16 (fp32 accumulate),
// fp32 online softmax. Operands pre-split/pre-swizzled; all LDS operand fetches
// are 128-bit (lane-contiguous runs of the 16-bit WMMA layout).

#define TB 256
constexpr int  Bc = 4, Sc = 4096, Ec = 1024, Ac = 64;
constexpr long BS = (long)Bc * Sc;   // 16384 flattened rows

typedef __attribute__((ext_vector_type(16))) __bf16 v16bf;
typedef __attribute__((ext_vector_type(8)))  float  v8f;

union BV {                 // 128-bit granular access to a 16-bit WMMA operand:
  v16bf v;                 // elements j<8  == K run [8h   .. 8h+7]
  uint4 q[2];              // elements j>=8 == K run [16+8h .. 16+8h+7]
};
union FV {                 // fp32 source for an A operand (same run structure)
  float  f[16];
  float4 q[4];
};

// fp32 -> bf16 hi/lo split: x ~= hi + lo, dropped term <= 2^-17 |x|
__device__ __forceinline__ void bsplit(float x, __bf16& hi, __bf16& lo) {
  hi = (__bf16)x;
  lo = (__bf16)(x - (float)hi);
}

// CDNA5 async memory->LDS copy (ASYNCcnt tracked), VFLAT GV form.
__device__ __forceinline__ void async_b128(const void* g, unsigned lds) {
  asm volatile("global_load_async_to_lds_b128 %0, %1, off"
               :: "v"(lds), "v"((unsigned long long)(size_t)g)
               : "memory");
}
__device__ __forceinline__ void wait_async0() {
  asm volatile("s_wait_asynccnt 0x0" ::: "memory");
}

// ---------------------------------------------------------------------------
// Kernel 0: split W{q,k,v} into bf16 hi/lo, pre-swizzled into the exact
// B-operand lane layout (two contiguous 8-element K-runs per lane).
// One thread per (o, nt, ks, lane): 3*4*32*32 = 12288 threads.
// ---------------------------------------------------------------------------
__global__ __launch_bounds__(TB) void wsplit_kernel(const float* __restrict__ Wq,
                                                    const float* __restrict__ Wk,
                                                    const float* __restrict__ Wv,
                                                    __bf16* __restrict__ wswh,
                                                    __bf16* __restrict__ wswl) {
  const int g = blockIdx.x * TB + threadIdx.x;   // 0..12287
  const int lane = g & 31, ks = (g >> 5) & 31, nt = (g >> 10) & 3, o = g >> 12;
  if (o >= 3) return;
  const float* W = (o == 0) ? Wq : ((o == 1) ? Wk : Wv);
  const int m = lane & 15, hh = lane >> 4;

  FV w;
  const float* wr = &W[(nt * 16 + m) * Ec + ks * 32 + 8 * hh];
  w.q[0] = *(const float4*)&wr[0];
  w.q[1] = *(const float4*)&wr[4];
  w.q[2] = *(const float4*)&wr[16];
  w.q[3] = *(const float4*)&wr[20];

  BV bh, bl;
#pragma unroll
  for (int j = 0; j < 16; ++j) {
    __bf16 hb, lb;
    bsplit(w.f[j], hb, lb);
    bh.v[j] = hb; bl.v[j] = lb;
  }
  const long idx = ((((long)o * 4 + nt) * 32 + ks) * 32 + lane) * 16;
  *(v16bf*)&wswh[idx] = bh.v;
  *(v16bf*)&wswl[idx] = bl.v;
}

// ---------------------------------------------------------------------------
// Kernel 1: q = (x Wq^T)/8 (fp32), k -> bf16 hi/lo [bs][64],
//           v -> bf16 hi/lo transposed [a][bs].
// x tile double-buffered in LDS via async-to-LDS; B operands are straight
// 32-byte vector loads of the pre-swizzled weights (zero VALU).
// ---------------------------------------------------------------------------
__global__ __launch_bounds__(TB) void qkv_proj(const float* __restrict__ x,
                                               const __bf16* __restrict__ wswh,
                                               const __bf16* __restrict__ wswl,
                                               float* __restrict__ qo,
                                               __bf16* __restrict__ kh,
                                               __bf16* __restrict__ kl,
                                               __bf16* __restrict__ vth,
                                               __bf16* __restrict__ vtl) {
  __shared__ float xs[2][128 * 32];                    // 32 KB double buffer
  const int  tid  = threadIdx.x;
  const int  wave = tid >> 5;
  const int  lane = tid & 31;
  const int  n16  = lane & 15;
  const int  h    = lane >> 4;
  const long row0 = (long)blockIdx.x * 128;
  const unsigned ldsX = (unsigned)(size_t)&xs[0][0];

  v8f acc[3][4];
#pragma unroll
  for (int o = 0; o < 3; ++o)
#pragma unroll
    for (int nt = 0; nt < 4; ++nt)
#pragma unroll
      for (int i = 0; i < 8; ++i) acc[o][nt][i] = 0.f;

  // stage helper: 128 rows x 32 floats = 16 KB, 64 B per thread (4x b128)
  const int srow = tid >> 1;
  const int scol = (tid & 1) * 16;
  {
    const float*  g = x + (row0 + srow) * Ec + scol;           // ks = 0
    const unsigned l = ldsX + (unsigned)(srow * 32 + scol) * 4u;
    async_b128(g,      l);      async_b128(g + 4,  l + 16);
    async_b128(g + 8,  l + 32); async_b128(g + 12, l + 48);
  }

  for (int ks = 0; ks < 32; ++ks) {
    wait_async0();
    __syncthreads();
    if (ks + 1 < 32) {                                          // stage ks+1
      const int ksn = ks + 1;
      const float*  g = x + (row0 + srow) * Ec + ksn * 32 + scol;
      const unsigned l = ldsX +
          (unsigned)(((ksn & 1) ? 4096 : 0) + srow * 32 + scol) * 4u;
      async_b128(g,      l);      async_b128(g + 4,  l + 16);
      async_b128(g + 8,  l + 32); async_b128(g + 12, l + 48);
    }

    // A operand (x tile): 4x ds_load_b128 + one split per element
    const float* xb = &xs[ks & 1][wave * 16 * 32 + n16 * 32 + 8 * h];
    FV xa;
    xa.q[0] = *(const float4*)&xb[0];
    xa.q[1] = *(const float4*)&xb[4];
    xa.q[2] = *(const float4*)&xb[16];
    xa.q[3] = *(const float4*)&xb[20];
    v16bf a_hi, a_lo;
#pragma unroll
    for (int j = 0; j < 16; ++j) {
      __bf16 hb, lb;
      bsplit(xa.f[j], hb, lb);
      a_hi[j] = hb; a_lo[j] = lb;
    }

#pragma unroll
    for (int o = 0; o < 3; ++o) {
#pragma unroll
      for (int nt = 0; nt < 4; ++nt) {
        const long bidx = ((((long)o * 4 + nt) * 32 + ks) * 32 + lane) * 16;
        const v16bf b_hi = *(const v16bf*)&wswh[bidx];   // 2x global_load_b128
        const v16bf b_lo = *(const v16bf*)&wswl[bidx];
        acc[o][nt] = __builtin_amdgcn_wmma_f32_16x16x32_bf16(false, a_hi, false, b_hi, (short)0, acc[o][nt], false, false);
        acc[o][nt] = __builtin_amdgcn_wmma_f32_16x16x32_bf16(false, a_hi, false, b_lo, (short)0, acc[o][nt], false, false);
        acc[o][nt] = __builtin_amdgcn_wmma_f32_16x16x32_bf16(false, a_lo, false, b_hi, (short)0, acc[o][nt], false, false);
      }
    }
  }

  // epilogue (C layout: row = r + 8h, col = nt*16 + n16)
  const long rowbase = row0 + wave * 16 + 8 * h;
#pragma unroll
  for (int nt = 0; nt < 4; ++nt) {
    const int col = nt * 16 + n16;
    union { __bf16 b[8]; uint4 u; } sh, sl;
#pragma unroll
    for (int r = 0; r < 8; ++r) {
      const long rowi = rowbase + r;
      qo[rowi * Ac + col] = acc[0][nt][r] * 0.125f;     // fold 1/sqrt(A)
      __bf16 hb, lb;
      bsplit(acc[1][nt][r], hb, lb);                    // k: [bs][64]
      kh[rowi * Ac + col] = hb;
      kl[rowi * Ac + col] = lb;
      bsplit(acc[2][nt][r], hb, lb);                    // v: transposed [a][bs]
      sh.b[r] = hb; sl.b[r] = lb;
    }
    *(uint4*)&vth[(long)col * BS + rowbase] = sh.u;     // 8 consecutive rows
    *(uint4*)&vtl[(long)col * BS + rowbase] = sl.u;
  }
}

// ---------------------------------------------------------------------------
// Kernel 2: causal flash attention. Block = 128 q-rows (8 waves x 16).
// K/V tiles (already bf16 hi/lo) stream into double-buffered LDS via
// async-to-LDS; P transposes through per-wave LDS scratch. All WMMA operand
// fetches are 2x b128 per v16bf. Wave-uniform fast path skips causal masking
// for tiles strictly below the diagonal.
// ---------------------------------------------------------------------------
__global__ __launch_bounds__(TB) void flash_attn(const float* __restrict__ q,
                                                 const __bf16* __restrict__ kh,
                                                 const __bf16* __restrict__ kl,
                                                 const __bf16* __restrict__ vth,
                                                 const __bf16* __restrict__ vtl,
                                                 float* __restrict__ out) {
  __shared__ __bf16 KhS[2][64 * 64], KlS[2][64 * 64];   // 32 KB
  __shared__ __bf16 VhS[2][64 * 64], VlS[2][64 * 64];   // 32 KB ([a][kv])
  __shared__ __bf16 Ph[8][16 * 64], Pl[8][16 * 64];     // 32 KB per-wave P

  const int  tid  = threadIdx.x;
  const int  wave = tid >> 5;
  const int  lane = tid & 31;
  const int  n16  = lane & 15;
  const int  h    = lane >> 4;
  const long row0 = (long)blockIdx.x * 128;
  const int  b    = (int)(row0 / Sc);
  const int  qs   = (int)(row0 % Sc);
  const long kvbase = (long)b * Sc;

  const unsigned lKh = (unsigned)(size_t)&KhS[0][0];
  const unsigned lKl = (unsigned)(size_t)&KlS[0][0];
  const unsigned lVh = (unsigned)(size_t)&VhS[0][0];
  const unsigned lVl = (unsigned)(size_t)&VlS[0][0];

  // Q A-operand hoisted (row = n16 per A layout), split once
  v16bf qa_hi[2], qa_lo[2];
  {
    const long qrow = row0 + wave * 16 + n16;
#pragma unroll
    for (int ks2 = 0; ks2 < 2; ++ks2) {
      const float* qr = &q[qrow * Ac + ks2 * 32 + 8 * h];
      FV qa;
      qa.q[0] = *(const float4*)&qr[0];
      qa.q[1] = *(const float4*)&qr[4];
      qa.q[2] = *(const float4*)&qr[16];
      qa.q[3] = *(const float4*)&qr[20];
#pragma unroll
      for (int j = 0; j < 16; ++j) {
        __bf16 hb, lb;
        bsplit(qa.f[j], hb, lb);
        qa_hi[ks2][j] = hb; qa_lo[ks2][j] = lb;
      }
    }
  }

  float mrow[8], lrow[8];
  v8f   oacc[4];
#pragma unroll
  for (int r = 0; r < 8; ++r) { mrow[r] = -1e30f; lrow[r] = 0.f; }
#pragma unroll
  for (int nt = 0; nt < 4; ++nt)
#pragma unroll
    for (int i = 0; i < 8; ++i) oacc[nt][i] = 0.f;

  const int nTiles = qs / 64 + 2;                       // causal upper bound

  // async stage of kv tile tt into buffer tt&1 (8x b128 per thread, 32 KB/tile)
  const int  var = tid >> 2;                            // V: 0..63 (a row)
  const int  vcc = (tid & 3) * 16;                      // V: col chunk (elems)
#define STAGE_KV(tt)                                                          \
  {                                                                           \
    const unsigned bo = ((tt) & 1) ? 8192u : 0u;                              \
    const long kb = (kvbase + (long)(tt) * 64) * 64 + tid * 16;               \
    async_b128(kh + kb,     lKh + bo + tid * 32);                             \
    async_b128(kh + kb + 8, lKh + bo + tid * 32 + 16);                        \
    async_b128(kl + kb,     lKl + bo + tid * 32);                             \
    async_b128(kl + kb + 8, lKl + bo + tid * 32 + 16);                        \
    const long vb = (long)var * BS + kvbase + (long)(tt) * 64 + vcc;          \
    const unsigned vlo = bo + (unsigned)(var * 64 + vcc) * 2u;                \
    async_b128(vth + vb,     lVh + vlo);                                      \
    async_b128(vth + vb + 8, lVh + vlo + 16);                                 \
    async_b128(vtl + vb,     lVl + vlo);                                      \
    async_b128(vtl + vb + 8, lVl + vlo + 16);                                 \
  }

  STAGE_KV(0);

  for (int t = 0; t < nTiles; ++t) {
    wait_async0();            // my tile-t copies are done
    __syncthreads();          // everyone's are done; buffer (t+1)&1 is free
    if (t + 1 < nTiles) STAGE_KV(t + 1);

    const __bf16* KhB = &KhS[t & 1][0];
    const __bf16* KlB = &KlS[t & 1][0];
    const __bf16* VhB = &VhS[t & 1][0];
    const __bf16* VlB = &VlS[t & 1][0];

    // ---- S = Q K^T (bf16x3) ----------------------------------------------
    v8f sacc[4];
#pragma unroll
    for (int nt = 0; nt < 4; ++nt)
#pragma unroll
      for (int i = 0; i < 8; ++i) sacc[nt][i] = 0.f;

#pragma unroll
    for (int ks2 = 0; ks2 < 2; ++ks2) {
#pragma unroll
      for (int nt = 0; nt < 4; ++nt) {
        BV bh, bl;
        const int kbase = (nt * 16 + n16) * 64 + ks2 * 32 + 8 * h;
        bh.q[0] = *(const uint4*)&KhB[kbase];
        bh.q[1] = *(const uint4*)&KhB[kbase + 16];
        bl.q[0] = *(const uint4*)&KlB[kbase];
        bl.q[1] = *(const uint4*)&KlB[kbase + 16];
        sacc[nt] = __builtin_amdgcn_wmma_f32_16x16x32_bf16(false, qa_hi[ks2], false, bh.v, (short)0, sacc[nt], false, false);
        sacc[nt] = __builtin_amdgcn_wmma_f32_16x16x32_bf16(false, qa_hi[ks2], false, bl.v, (short)0, sacc[nt], false, false);
        sacc[nt] = __builtin_amdgcn_wmma_f32_16x16x32_bf16(false, qa_lo[ks2], false, bh.v, (short)0, sacc[nt], false, false);
      }
    }

    // ---- causal mask: only tiles touching this wave's diagonal ------------
    const int rowb = qs + wave * 16 + 8 * h;            // lane's base q-row
    if (t * 64 + 63 > qs + wave * 16) {                 // wave-uniform branch
#pragma unroll
      for (int r = 0; r < 8; ++r)
#pragma unroll
        for (int nt = 0; nt < 4; ++nt) {
          const int kvcol = t * 64 + nt * 16 + n16;
          if (kvcol > rowb + r) sacc[nt][r] = -1e30f;
        }
    }

    // ---- online softmax (fp32) -------------------------------------------
#pragma unroll
    for (int r = 0; r < 8; ++r) {
      float mx = -1e30f;
#pragma unroll
      for (int nt = 0; nt < 4; ++nt) mx = fmaxf(mx, sacc[nt][r]);
#pragma unroll
      for (int d = 1; d < 16; d <<= 1) mx = fmaxf(mx, __shfl_xor(mx, d, 32));
      const float mnew = fmaxf(mrow[r], mx);
      const float corr = __expf(mrow[r] - mnew);
      mrow[r] = mnew;
      float psum = 0.f;
#pragma unroll
      for (int nt = 0; nt < 4; ++nt) {
        const float p = __expf(sacc[nt][r] - mnew);
        psum += p;
        __bf16 hh2, ll2;
        bsplit(p, hh2, ll2);
        Ph[wave][(8*h + r) * 64 + nt*16 + n16] = hh2;
        Pl[wave][(8*h + r) * 64 + nt*16 + n16] = ll2;
      }
#pragma unroll
      for (int d = 1; d < 16; d <<= 1) psum += __shfl_xor(psum, d, 32);
      lrow[r] = lrow[r] * corr + psum;
#pragma unroll
      for (int nt = 0; nt < 4; ++nt) oacc[nt][r] *= corr;
    }

    // ---- O += P V (bf16x3); P read back in A layout, V from [a][kv] -------
#pragma unroll
    for (int ks2 = 0; ks2 < 2; ++ks2) {
      BV pah, pal;
      const int pbase = n16 * 64 + ks2 * 32 + 8 * h;
      pah.q[0] = *(const uint4*)&Ph[wave][pbase];
      pah.q[1] = *(const uint4*)&Ph[wave][pbase + 16];
      pal.q[0] = *(const uint4*)&Pl[wave][pbase];
      pal.q[1] = *(const uint4*)&Pl[wave][pbase + 16];
#pragma unroll
      for (int nt = 0; nt < 4; ++nt) {
        BV bh, bl;
        const int vbase = (nt * 16 + n16) * 64 + ks2 * 32 + 8 * h;
        bh.q[0] = *(const uint4*)&VhB[vbase];
        bh.q[1] = *(const uint4*)&VhB[vbase + 16];
        bl.q[0] = *(const uint4*)&VlB[vbase];
        bl.q[1] = *(const uint4*)&VlB[vbase + 16];
        oacc[nt] = __builtin_amdgcn_wmma_f32_16x16x32_bf16(false, pah.v, false, bh.v, (short)0, oacc[nt], false, false);
        oacc[nt] = __builtin_amdgcn_wmma_f32_16x16x32_bf16(false, pah.v, false, bl.v, (short)0, oacc[nt], false, false);
        oacc[nt] = __builtin_amdgcn_wmma_f32_16x16x32_bf16(false, pal.v, false, bh.v, (short)0, oacc[nt], false, false);
      }
    }
  }
#undef STAGE_KV

  // epilogue: normalize and round to 4 decimals (RNE == jnp.round)
#pragma unroll
  for (int nt = 0; nt < 4; ++nt)
#pragma unroll
    for (int r = 0; r < 8; ++r) {
      const long rowi = row0 + wave * 16 + 8 * h + r;
      float val = oacc[nt][r] / lrow[r];
      val = rintf(val * 1.0e4f) * 1.0e-4f;
      out[rowi * Ac + nt * 16 + n16] = val;
    }
}

// ---------------------------------------------------------------------------
extern "C" void kernel_launch(void* const* d_in, const int* in_sizes, int n_in,
                              void* d_out, int out_size, void* d_ws, size_t ws_size,
                              hipStream_t stream) {
  (void)in_sizes; (void)n_in; (void)out_size; (void)ws_size;
  const float* x  = (const float*)d_in[0];   // [B,S,E]
  const float* Wq = (const float*)d_in[1];   // [A,E]
  const float* Wk = (const float*)d_in[2];
  const float* Wv = (const float*)d_in[3];

  char* w = (char*)d_ws;                     // ~12.75 MB of scratch
  float*  qb   = (float*)w;  w += BS * Ac * 4;          // q fp32 (pre-scaled)
  __bf16* khb  = (__bf16*)w; w += BS * Ac * 2;          // k hi   [bs][64]
  __bf16* klb  = (__bf16*)w; w += BS * Ac * 2;          // k lo
  __bf16* vthb = (__bf16*)w; w += BS * Ac * 2;          // v hi   [a][bs]
  __bf16* vtlb = (__bf16*)w; w += BS * Ac * 2;          // v lo
  __bf16* wswh = (__bf16*)w; w += 3L * 4 * 32 * 32 * 16 * 2;  // W hi swizzled
  __bf16* wswl = (__bf16*)w;                                   // W lo swizzled
  float* outp = (float*)d_out;               // [B,S,A] fp32

  wsplit_kernel<<<dim3(48),            dim3(TB), 0, stream>>>(Wq, Wk, Wv, wswh, wswl);
  qkv_proj     <<<dim3((int)(BS/128)), dim3(TB), 0, stream>>>(x, wswh, wswl, qb, khb, klb, vthb, vtlb);
  flash_attn   <<<dim3((int)(BS/128)), dim3(TB), 0, stream>>>(qb, khb, klb, vthb, vtlb, outp);
}
